// CrossAttention_71073118814901
// MI455X (gfx1250) — matrix operand order repaired and checked by reading, not compiled, not run on hardware
//
#include <hip/hip_runtime.h>
#include <hip/hip_bf16.h>

#define BATCH 4
#define CCH   256
#define NHW   4096
#define DQK   32

typedef _Float16 h16;
typedef __attribute__((ext_vector_type(16))) _Float16 v16h;
typedef __attribute__((ext_vector_type(8)))  float    v8f;
typedef __attribute__((ext_vector_type(4)))  float    f32x4;  // POD 16B vector

union V16H { v16h v; f32x4 f4[2]; };

// ---------------- Projection: q = Wq x1, k = Wk x2 (d=32 outputs) ----------------
__global__ __launch_bounds__(256) void proj_qk(
    const float* __restrict__ x1, const float* __restrict__ x2,
    const float* __restrict__ wq1, const float* __restrict__ bq1,
    const float* __restrict__ wk1, const float* __restrict__ bk1,
    const float* __restrict__ wq2, const float* __restrict__ bq2,
    const float* __restrict__ wk2, const float* __restrict__ bk2,
    h16* __restrict__ Qh, h16* __restrict__ Kh)
{
    const int dd = threadIdx.x & 31;       // output channel 0..31
    const int ii = threadIdx.x >> 5;       // 8 spatial positions per block
    int t = blockIdx.x;
    const int ich   = t % (NHW / 8); t /= (NHW / 8);
    const int which = t & 1;         t >>= 1;   // 0 = q (from x1), 1 = k (from x2)
    const int r     = t & 1;                    // branch
    const int b     = t >> 1;                   // batch
    const int i     = ich * 8 + ii;

    const float* x    = which ? x2 : x1;
    const float* w    = which ? (r ? wk2 : wk1) : (r ? wq2 : wq1);
    const float* bias = which ? (r ? bk2 : bk1) : (r ? bq2 : bq1);

    float acc = bias[dd];
    const float* xp = x + (size_t)b * CCH * NHW + i;
    const float* wp = w + dd * CCH;
    #pragma unroll 8
    for (int tt = 0; tt < CCH; ++tt)
        acc += wp[tt] * xp[(size_t)tt * NHW];

    h16* dst = which ? Kh : Qh;
    dst[(((size_t)(b * 2 + r)) * NHW + i) * DQK + dd] = (h16)acc;
}

// ---------------- Projection: v = Wv x2 (256 outputs), stored [c][j] (j contiguous) ----------------
__global__ __launch_bounds__(256) void proj_v(
    const float* __restrict__ x2,
    const float* __restrict__ wv1, const float* __restrict__ bv1,
    const float* __restrict__ wv2, const float* __restrict__ bv2,
    h16* __restrict__ Vh)
{
    const int j = (blockIdx.x % (NHW / 256)) * 256 + threadIdx.x;
    int t = blockIdx.x / (NHW / 256);
    const int cg = t % 16; t /= 16;
    const int r  = t & 1;
    const int b  = t >> 1;
    const int c0 = cg * 16;

    const float* w    = r ? wv2 : wv1;
    const float* bias = r ? bv2 : bv1;

    float acc[16];
    #pragma unroll
    for (int cc = 0; cc < 16; ++cc) acc[cc] = bias[c0 + cc];

    const float* xp = x2 + (size_t)b * CCH * NHW + j;
    for (int tt = 0; tt < CCH; ++tt) {
        const float xv = xp[(size_t)tt * NHW];          // coalesced across block
        #pragma unroll
        for (int cc = 0; cc < 16; ++cc)
            acc[cc] += w[(c0 + cc) * CCH + tt] * xv;    // scalar (uniform) loads
    }
    #pragma unroll
    for (int cc = 0; cc < 16; ++cc)
        Vh[(((size_t)(b * 2 + r)) * CCH + c0 + cc) * NHW + j] = (h16)acc[cc];
}

// ---------------- Fused flash attention over both branches + residual ----------------
// grid = (N/16, B), block = 64 (2 waves). Wave w owns channels [w*128, w*128+128).
//
// Trick: compute S^T = WMMA(A = K-tile, B = Q-tile). In the C/D layout each lane
// then holds a FIXED query index i = lane%16 and 8 consecutive j values per tile,
// which is exactly the A-operand layout needed for O += P * V^T:
//    pa[e] = (e < 8) ? exp(S1t[e]) : exp(S2t[e-8])    (purely lane-local)
// Softmax denominators accumulate per-lane (fixed i); one xor-16 shuffle + 8
// broadcasts per BRANCH finish them. No running max: softmax is shift-invariant
// and energies (d=32 dot of O(1)*O(0.3) values) are far below f32 exp overflow.
__global__ __launch_bounds__(64) void fa_fused(
    const h16* __restrict__ Qh, const h16* __restrict__ Kh, const h16* __restrict__ Vh,
    const float* __restrict__ x1, const float* __restrict__ x2,
    float* __restrict__ out)
{
    const int lane = threadIdx.x & 31;
    const int wave = threadIdx.x >> 5;
    const int half = lane >> 4;          // 0: lanes 0-15, 1: lanes 16-31
    const int lr   = lane & 15;
    const int i0   = blockIdx.x * 16;
    const int b    = blockIdx.y;
    const int c0   = wave * 128;

    for (int r = 0; r < 2; ++r) {
        const size_t br = (size_t)(b * 2 + r);

        // B operand: Q tile (32x16, K=d). Lane holds column i = i0+lr, K values
        // half*8..+7 and 16+half*8..+7. Loaded once per branch.
        V16H bq;
        {
            const h16* qb = Qh + (br * NHW + i0 + lr) * DQK + half * 8;
            bq.f4[0] = *(const f32x4*)(qb);
            bq.f4[1] = *(const f32x4*)(qb + 16);
        }

        v8f O[8];
        #pragma unroll
        for (int ct = 0; ct < 8; ++ct)
            #pragma unroll
            for (int e = 0; e < 8; ++e) O[ct][e] = 0.0f;
        float lsum = 0.0f;                 // per-lane partial denominator for i = i0+lr

        const h16* Kb = Kh + br * NHW * DQK;
        const h16* Vb = Vh + br * CCH * NHW;

        for (int j = 0; j < NHW; j += 32) {
            // A operands: K rows j+lr and j+16+lr (lane<16: K-dims 0-7 & 16-23; lane>=16: 8-15 & 24-31).
            V16H k1, k2;
            const h16* kp1 = Kb + (size_t)(j + lr) * DQK + half * 8;
            k1.f4[0] = *(const f32x4*)(kp1);
            k1.f4[1] = *(const f32x4*)(kp1 + 16);
            const h16* kp2 = Kb + (size_t)(j + 16 + lr) * DQK + half * 8;
            k2.f4[0] = *(const f32x4*)(kp2);
            k2.f4[1] = *(const f32x4*)(kp2 + 16);

            v8f zero;
            #pragma unroll
            for (int e = 0; e < 8; ++e) zero[e] = 0.0f;

            // S^T tiles: element g = S[i0+lr][j + half*8 + g] (tile2: +16).
            v8f S1t = __builtin_amdgcn_wmma_f32_16x16x32_f16(false, k1.v, false, bq.v, (short)0, zero, false, false);
            v8f S2t = __builtin_amdgcn_wmma_f32_16x16x32_f16(false, k2.v, false, bq.v, (short)0, zero, false, false);

            // Unnormalized probabilities; build PV A-operand in-lane.
            V16H pa;
            float ls = 0.0f;
            #pragma unroll
            for (int g = 0; g < 8; ++g) {
                const float p1 = __expf(S1t[g]);
                const float p2 = __expf(S2t[g]);
                ls += p1 + p2;
                pa.v[g]     = (h16)p1;
                pa.v[g + 8] = (h16)p2;
            }
            lsum += ls;

            // O += P (16x32) * V^T (32x16) for 8 channel tiles.
            #pragma unroll
            for (int ct = 0; ct < 8; ++ct) {
                V16H bv;
                const h16* vp = Vb + (size_t)(c0 + ct * 16 + lr) * NHW + j + half * 8;
                bv.f4[0] = *(const f32x4*)(vp);
                bv.f4[1] = *(const f32x4*)(vp + 16);
                O[ct] = __builtin_amdgcn_wmma_f32_16x16x32_f16(false, pa.v, false, bv.v, (short)0, O[ct], false, false);
            }
        }

        // Finish denominators: total over j for i = i0+lr, then broadcast to the
        // element-indexed rows of the O tiles (row of element g is g + half*8).
        const float lt = lsum + __shfl_xor(lsum, 16, 32);
        float rl[8];
        #pragma unroll
        for (int g = 0; g < 8; ++g)
            rl[g] = 1.0f / __shfl(lt, g + half * 8, 32);

        if (r == 0) {
            #pragma unroll
            for (int ct = 0; ct < 8; ++ct)
                #pragma unroll
                for (int e = 0; e < 8; ++e) {
                    const int c = c0 + ct * 16 + lr;
                    const int i = i0 + e + half * 8;
                    out[((size_t)b * CCH + c) * NHW + i] = O[ct][e] * rl[e];
                }
        } else {
            #pragma unroll
            for (int ct = 0; ct < 8; ++ct)
                #pragma unroll
                for (int e = 0; e < 8; ++e) {
                    const int c = c0 + ct * 16 + lr;
                    const int i = i0 + e + half * 8;
                    const size_t idx = ((size_t)b * CCH + c) * NHW + i;
                    out[idx] = O[ct][e] * rl[e] + out[idx] + x1[idx] + x2[idx];
                }
        }
    }
}

extern "C" void kernel_launch(void* const* d_in, const int* in_sizes, int n_in,
                              void* d_out, int out_size, void* d_ws, size_t ws_size,
                              hipStream_t stream) {
    (void)in_sizes; (void)n_in; (void)out_size; (void)ws_size;
    const float* x1  = (const float*)d_in[0];
    const float* x2  = (const float*)d_in[1];
    const float* wq1 = (const float*)d_in[2];
    const float* bq1 = (const float*)d_in[3];
    const float* wk1 = (const float*)d_in[4];
    const float* bk1 = (const float*)d_in[5];
    const float* wv1 = (const float*)d_in[6];
    const float* bv1 = (const float*)d_in[7];
    const float* wq2 = (const float*)d_in[8];
    const float* bq2 = (const float*)d_in[9];
    const float* wk2 = (const float*)d_in[10];
    const float* bk2 = (const float*)d_in[11];
    const float* wv2 = (const float*)d_in[12];
    const float* bv2 = (const float*)d_in[13];
    float* out = (float*)d_out;

    // Workspace: Q (2 MB) | K (2 MB) | V (16 MB), fp16.
    h16* Qh = (h16*)d_ws;
    h16* Kh = Qh + (size_t)BATCH * 2 * NHW * DQK;
    h16* Vh = Kh + (size_t)BATCH * 2 * NHW * DQK;

    proj_qk<<<dim3(BATCH * 2 * 2 * (NHW / 8)), 256, 0, stream>>>(
        x1, x2, wq1, bq1, wk1, bk1, wq2, bq2, wk2, bk2, Qh, Kh);
    proj_v<<<dim3(BATCH * 2 * 16 * (NHW / 256)), 256, 0, stream>>>(
        x2, wv1, bv1, wv2, bv2, Vh);
    fa_fused<<<dim3(NHW / 16, BATCH), 64, 0, stream>>>(
        Qh, Kh, Vh, x1, x2, out);
}
